// PyramidAttention_18794776888055
// MI455X (gfx1250) — matrix operand order repaired and checked by reading, not compiled, not run on hardware
//
#include <hip/hip_runtime.h>
#include <math.h>

typedef _Float16 f16;
typedef __attribute__((ext_vector_type(16))) _Float16 v16h;
typedef __attribute__((ext_vector_type(8)))  _Float16 v8h;
typedef __attribute__((ext_vector_type(8)))  float    v8f;

#define B_    2
#define C_    50
#define CR_   25
#define H_    48
#define W_    48
#define HW_   2304
#define PTOT  7470
#define LPAD  7488
#define DPAD  256
#define VPAD  512
#define NLEV  5
#define KTILES (LPAD/32)

__device__ __constant__ int g_lev_off[NLEV+1] = {0, 2304, 4153, 5597, 6686, 7470};
__device__ __constant__ int g_lev_dim[NLEV]   = {48, 43, 38, 33, 28};

__device__ __forceinline__ int find_lev(int p) {
    int lv = 0;
#pragma unroll
    for (int i = 1; i < NLEV; i++) if (p >= g_lev_off[i]) lv = i;
    return lv;
}

__device__ __forceinline__ float cubw(float t) {
    const float a = -0.75f;
    float at = fabsf(t);
    float w1 = ((a + 2.0f) * at - (a + 3.0f)) * at * at + 1.0f;
    float w2 = (((at - 5.0f) * at + 8.0f) * at - 4.0f) * a;
    return (at <= 1.0f) ? w1 : ((at < 2.0f) ? w2 : 0.0f);
}

// ---------------- Kernel 1: bicubic pyramid of x (level0 = copy) ----------------
__global__ void k_pyramid(const float* __restrict__ x, float* __restrict__ pyr) {
    int gid = blockIdx.x * blockDim.x + threadIdx.x;
    const int total = B_ * C_ * PTOT;
    if (gid >= total) return;
    int p  = gid % PTOT;
    int bc = gid / PTOT;
    int lv = find_lev(p);
    int d  = g_lev_dim[lv];
    int rp = p - g_lev_off[lv];
    int oy = rp / d, ox = rp % d;
    float v;
    if (lv == 0) {
        v = x[(size_t)bc * HW_ + oy * W_ + ox];
    } else {
        float sc = 47.0f / (float)(d - 1);
        float sy = (float)oy * sc, sx = (float)ox * sc;
        float fy = floorf(sy), fx = floorf(sx);
        int   iy = (int)fy,    ix = (int)fx;
        float ty = sy - fy,    tx = sx - fx;
        float wy[4], wx[4]; int yy[4], xx[4];
#pragma unroll
        for (int k = 0; k < 4; k++) {
            int dd = k - 1;
            wy[k] = cubw(ty - (float)dd);
            wx[k] = cubw(tx - (float)dd);
            int a0 = iy + dd; a0 = a0 < 0 ? 0 : (a0 > 47 ? 47 : a0); yy[k] = a0;
            int a1 = ix + dd; a1 = a1 < 0 ? 0 : (a1 > 47 ? 47 : a1); xx[k] = a1;
        }
        v = 0.0f;
#pragma unroll
        for (int ky = 0; ky < 4; ky++)
#pragma unroll
            for (int kx = 0; kx < 4; kx++)
                v += wy[ky] * wx[kx] * x[(size_t)bc * HW_ + yy[ky] * W_ + xx[kx]];
    }
    pyr[(size_t)bc * PTOT + p] = v;
}

// ---------------- Kernel 2: conv1x1 + PReLU ----------------
__global__ void k_conv1x1(const float* __restrict__ in, const float* __restrict__ Wt,
                          const float* __restrict__ bias, const float* __restrict__ a,
                          float* __restrict__ out, int O, int P, int total) {
    int gid = blockIdx.x * blockDim.x + threadIdx.x;
    if (gid >= total) return;
    int pid = gid % P;
    int bo  = gid / P;
    int o   = bo % O;
    int b   = bo / O;
    const float* ip = in + (size_t)b * C_ * P + pid;
    const float* wp = Wt + o * C_;
    float s = bias[o];
#pragma unroll 5
    for (int c = 0; c < C_; c++) s += wp[c] * ip[(size_t)c * P];
    float al = a[0];
    out[gid] = (s >= 0.0f) ? s : al * s;
}

// ---------------- Kernel 3: normalized key patches -> f16 K [B][LPAD][DPAD] ----------------
__global__ void k_keys(const float* __restrict__ mp, f16* __restrict__ Km) {
    int gid = blockIdx.x * blockDim.x + threadIdx.x;
    if (gid >= B_ * LPAD) return;
    int l = gid % LPAD, b = gid / LPAD;
    f16* kr = Km + (size_t)gid * DPAD;
    if (l >= PTOT) {
        for (int i = 0; i < DPAD; i++) kr[i] = (f16)0.0f;
        return;
    }
    int lv = find_lev(l);
    int d  = g_lev_dim[lv];
    int rp = l - g_lev_off[lv];
    int y = rp / d, x = rp % d;
    const float* mbase = mp + (size_t)b * CR_ * PTOT;
    float ss = 0.0f;
    for (int c = 0; c < CR_; c++) {
        const float* cp = mbase + (size_t)c * PTOT + g_lev_off[lv];
#pragma unroll
        for (int di = 0; di < 3; di++)
#pragma unroll
            for (int dj = 0; dj < 3; dj++) {
                int yy = y + di - 1, xx = x + dj - 1;
                float v = (yy >= 0 && yy < d && xx >= 0 && xx < d) ? cp[yy * d + xx] : 0.0f;
                ss += v * v;
            }
    }
    float sc = 10.0f / fmaxf(sqrtf(ss), 1e-4f);
    for (int c = 0; c < CR_; c++) {
        const float* cp = mbase + (size_t)c * PTOT + g_lev_off[lv];
#pragma unroll
        for (int di = 0; di < 3; di++)
#pragma unroll
            for (int dj = 0; dj < 3; dj++) {
                int yy = y + di - 1, xx = x + dj - 1;
                float v = (yy >= 0 && yy < d && xx >= 0 && xx < d) ? cp[yy * d + xx] : 0.0f;
                kr[c * 9 + di * 3 + dj] = (f16)(v * sc);
            }
    }
    for (int i = 225; i < DPAD; i++) kr[i] = (f16)0.0f;
}

// ---------------- Kernel 4: query patches -> f16 Q [B][HW][DPAD] ----------------
__global__ void k_queries(const float* __restrict__ mb, f16* __restrict__ Qm) {
    int gid = blockIdx.x * blockDim.x + threadIdx.x;
    if (gid >= B_ * HW_) return;
    int p = gid % HW_, b = gid / HW_;
    int y = p / W_, x = p % W_;
    f16* qr = Qm + (size_t)gid * DPAD;
    const float* base = mb + (size_t)b * CR_ * HW_;
    for (int c = 0; c < CR_; c++) {
        const float* cp = base + (size_t)c * HW_;
#pragma unroll
        for (int di = 0; di < 3; di++)
#pragma unroll
            for (int dj = 0; dj < 3; dj++) {
                int yy = y + di - 1, xx = x + dj - 1;
                float v = (yy >= 0 && yy < H_ && xx >= 0 && xx < W_) ? cp[yy * W_ + xx] : 0.0f;
                qr[c * 9 + di * 3 + dj] = (f16)v;
            }
    }
    for (int i = 225; i < DPAD; i++) qr[i] = (f16)0.0f;
}

// ---------------- Kernel 5: transposed value matrix Vt [B][VPAD][LPAD] f16 ----------------
__global__ void k_vt(const float* __restrict__ bp, f16* __restrict__ Vt) {
    int gid = blockIdx.x * blockDim.x + threadIdx.x;
    const int total = B_ * VPAD * LPAD;
    if (gid >= total) return;
    int l   = gid % LPAD;
    int t   = gid / LPAD;
    int col = t % VPAD;
    int b   = t / VPAD;
    f16 v = (f16)0.0f;
    if (col < 450 && l < PTOT) {
        int c = col / 9, tt = col % 9, i = tt / 3, j = tt % 3;
        int lv = find_lev(l);
        int d  = g_lev_dim[lv];
        int rp = l - g_lev_off[lv];
        int y = rp / d, x = rp % d;
        int sy = y + 1 - i, sx = x + 1 - j;
        if (sy >= 0 && sy < d && sx >= 0 && sx < d)
            v = (f16)bp[((size_t)b * C_ + c) * PTOT + g_lev_off[lv] + sy * d + sx];
    }
    Vt[gid] = v;
}

// ---------------- Kernel 6: fused flash attention ----------------
__global__ __launch_bounds__(256) void k_flash(const f16* __restrict__ Km,
                                               const f16* __restrict__ Qm,
                                               const f16* __restrict__ Vt,
                                               float* __restrict__ Gt) {
    __shared__ f16  Qs[16 * DPAD];
    __shared__ f16  Ps[16 * 32];
    __shared__ float corr_s[16];
    __shared__ float sfin[16];

    const int b     = blockIdx.y;
    const int qbase = blockIdx.x * 16;
    const int tid   = threadIdx.x;
    const int wave  = tid >> 5;
    const int lane  = tid & 31;
    const int nlo   = lane & 15;
    const int khi   = lane >> 4;
    const int moff  = khi * 8;

    {
        const v8h* src = (const v8h*)(Qm + ((size_t)(b * HW_ + qbase)) * DPAD);
        v8h* dst = (v8h*)Qs;
        for (int i = tid; i < (16 * DPAD) / 8; i += 256) dst[i] = src[i];
    }
    __syncthreads();

    v16h Qb[8];
    if (wave == 0) {
#pragma unroll
        for (int ch = 0; ch < 8; ch++) {
            const f16* p = &Qs[nlo * DPAD + ch * 32 + 16 * khi];
#pragma unroll
            for (int i = 0; i < 16; i++) Qb[ch][i] = p[i];
        }
    }

    float m_prev = -__builtin_inff();
    float s_run  = 0.0f;
    v8f acc[4];
#pragma unroll
    for (int vt = 0; vt < 4; vt++)
#pragma unroll
        for (int r = 0; r < 8; r++) acc[vt][r] = 0.0f;

    for (int t = 0; t < KTILES; t++) {
        const int kbase = t * 32;
        if (wave == 0) {
            v8f S[2];
#pragma unroll
            for (int hf = 0; hf < 2; hf++) {
                v8f a;
#pragma unroll
                for (int r = 0; r < 8; r++) a[r] = 0.0f;
                const f16* krow = Km + ((size_t)(b * LPAD + kbase + hf * 16 + nlo)) * DPAD;
#pragma unroll
                for (int ch = 0; ch < 8; ch++) {
                    const f16* kp = krow + ch * 32;
                    v8h lo = *(const v8h*)(kp + 16 * khi);
                    v8h hi = *(const v8h*)(kp + 16 * khi + 8);
                    v16h A;
#pragma unroll
                    for (int i = 0; i < 8; i++) { A[i] = lo[i]; A[8 + i] = hi[i]; }
                    a = __builtin_amdgcn_wmma_f32_16x16x32_f16(false, A, false, Qb[ch],
                                                               (short)0, a, false, false);
                }
                S[hf] = a;
            }
            if (t + 1 < KTILES)
                __builtin_prefetch(Km + ((size_t)(b * LPAD + kbase + 32 + nlo)) * DPAD, 0, 1);

            float mloc = -__builtin_inff();
#pragma unroll
            for (int r = 0; r < 8; r++) {
                if (kbase + moff + r      < PTOT) mloc = fmaxf(mloc, S[0][r]);
                if (kbase + 16 + moff + r < PTOT) mloc = fmaxf(mloc, S[1][r]);
            }
            mloc = fmaxf(mloc, __shfl_xor(mloc, 16, 32));
            float m_new = fmaxf(m_prev, mloc);
            float p0[8], p1[8];
            float ssum = 0.0f;
#pragma unroll
            for (int r = 0; r < 8; r++) {
                p0[r] = (kbase + moff + r      < PTOT) ? __expf(S[0][r] - m_new) : 0.0f;
                p1[r] = (kbase + 16 + moff + r < PTOT) ? __expf(S[1][r] - m_new) : 0.0f;
                ssum += p0[r] + p1[r];
            }
            ssum += __shfl_xor(ssum, 16, 32);
            float cf = __expf(m_prev - m_new);
            s_run = s_run * cf + ssum;
            m_prev = m_new;
            if (lane < 16) corr_s[lane] = cf;
#pragma unroll
            for (int r = 0; r < 8; r++) {
                Ps[nlo * 32 + moff + r]      = (f16)p0[r];
                Ps[nlo * 32 + 16 + moff + r] = (f16)p1[r];
            }
        }
        __syncthreads();

        float cf8[8];
#pragma unroll
        for (int r = 0; r < 8; r++) cf8[r] = corr_s[moff + r];
        v16h Pa;
        {
            const f16* pp = &Ps[nlo * 32 + 8 * khi];
#pragma unroll
            for (int i = 0; i < 8; i++) { Pa[i] = pp[i]; Pa[8 + i] = pp[16 + i]; }
        }
#pragma unroll
        for (int vt = 0; vt < 4; vt++) {
            const int col = (wave * 4 + vt) * 16 + nlo;
            const f16* vp = Vt + ((size_t)(b * VPAD + col)) * LPAD + kbase + 8 * khi;
            v8h lo = *(const v8h*)vp;
            v8h hi = *(const v8h*)(vp + 16);
            v16h Bv;
#pragma unroll
            for (int i = 0; i < 8; i++) { Bv[i] = lo[i]; Bv[8 + i] = hi[i]; }
            v8f c = acc[vt];
#pragma unroll
            for (int r = 0; r < 8; r++) c[r] *= cf8[r];
            acc[vt] = __builtin_amdgcn_wmma_f32_16x16x32_f16(false, Pa, false, Bv,
                                                             (short)0, c, false, false);
        }
        __syncthreads();
    }

    if (wave == 0 && lane < 16) sfin[lane] = s_run;
    __syncthreads();
    float inv8[8];
#pragma unroll
    for (int r = 0; r < 8; r++) inv8[r] = 1.0f / sfin[moff + r];
#pragma unroll
    for (int vt = 0; vt < 4; vt++) {
        const int col = (wave * 4 + vt) * 16 + nlo;
        float* gp = Gt + ((size_t)(b * VPAD + col)) * HW_ + qbase + moff;
#pragma unroll
        for (int r = 0; r < 8; r++) gp[r] = acc[vt][r] * inv8[r];
    }
}

// ---------------- Kernel 7: 9-tap gather + residual ----------------
__global__ void k_out(const float* __restrict__ x, const float* __restrict__ Gt,
                      float* __restrict__ out) {
    int gid = blockIdx.x * blockDim.x + threadIdx.x;
    const int total = B_ * C_ * HW_;
    if (gid >= total) return;
    int pw = gid % W_;
    int ph = (gid / W_) % H_;
    int c  = (gid / HW_) % C_;
    int b  = gid / (C_ * HW_);
    float s = 0.0f;
#pragma unroll
    for (int i = 0; i < 3; i++)
#pragma unroll
        for (int j = 0; j < 3; j++) {
            int hh = ph + i - 1, ww = pw + j - 1;
            if (hh >= 0 && hh < H_ && ww >= 0 && ww < W_)
                s += Gt[((size_t)(b * VPAD + c * 9 + i * 3 + j)) * HW_ + hh * W_ + ww];
        }
    out[gid] = x[gid] + 0.25f * s;
}

extern "C" void kernel_launch(void* const* d_in, const int* in_sizes, int n_in,
                              void* d_out, int out_size, void* d_ws, size_t ws_size,
                              hipStream_t stream) {
    const float* x    = (const float*)d_in[0];
    const float* W_mb = (const float*)d_in[1];
    const float* b_mb = (const float*)d_in[2];
    const float* a_mb = (const float*)d_in[3];
    const float* W_m  = (const float*)d_in[4];
    const float* b_m  = (const float*)d_in[5];
    const float* a_m  = (const float*)d_in[6];
    const float* W_a  = (const float*)d_in[7];
    const float* b_a  = (const float*)d_in[8];
    const float* a_a  = (const float*)d_in[9];
    (void)in_sizes; (void)n_in; (void)out_size; (void)ws_size;

    char* ws = (char*)d_ws;
    size_t off = 0;
    auto alloc = [&](size_t bytes) {
        void* p = ws + off;
        off = (off + bytes + 255) & ~(size_t)255;
        return p;
    };
    float* pyr  = (float*)alloc((size_t)B_ * C_  * PTOT * 4);
    float* mb   = (float*)alloc((size_t)B_ * CR_ * HW_  * 4);
    float* mpyr = (float*)alloc((size_t)B_ * CR_ * PTOT * 4);
    float* bpyr = (float*)alloc((size_t)B_ * C_  * PTOT * 4);
    f16*   Kmx  = (f16*)  alloc((size_t)B_ * LPAD * DPAD * 2);
    f16*   Qmx  = (f16*)  alloc((size_t)B_ * HW_  * DPAD * 2);
    f16*   Vtx  = (f16*)  alloc((size_t)B_ * VPAD * LPAD * 2);
    float* Gt   = (float*)alloc((size_t)B_ * VPAD * HW_  * 4);

    const int TB = 256;
    int tot = B_ * C_ * PTOT;
    k_pyramid<<<(tot + TB - 1) / TB, TB, 0, stream>>>(x, pyr);

    tot = B_ * CR_ * HW_;
    k_conv1x1<<<(tot + TB - 1) / TB, TB, 0, stream>>>(x, W_mb, b_mb, a_mb, mb, CR_, HW_, tot);
    tot = B_ * CR_ * PTOT;
    k_conv1x1<<<(tot + TB - 1) / TB, TB, 0, stream>>>(pyr, W_m, b_m, a_m, mpyr, CR_, PTOT, tot);
    tot = B_ * C_ * PTOT;
    k_conv1x1<<<(tot + TB - 1) / TB, TB, 0, stream>>>(pyr, W_a, b_a, a_a, bpyr, C_, PTOT, tot);

    tot = B_ * LPAD;
    k_keys<<<(tot + TB - 1) / TB, TB, 0, stream>>>(mpyr, Kmx);
    tot = B_ * HW_;
    k_queries<<<(tot + TB - 1) / TB, TB, 0, stream>>>(mb, Qmx);
    tot = B_ * VPAD * LPAD;
    k_vt<<<(tot + TB - 1) / TB, TB, 0, stream>>>(bpyr, Vtx);

    dim3 grid(HW_ / 16, B_);
    k_flash<<<grid, 256, 0, stream>>>(Kmx, Qmx, Vtx, Gt);

    tot = B_ * C_ * HW_;
    k_out<<<(tot + TB - 1) / TB, TB, 0, stream>>>(x, Gt, (float*)d_out);
}
